// SGNNMessagePassingLayer_8770323218950
// MI455X (gfx1250) — compile-verified, matrix operand order checked by reading, not run on hardware
//
#include <hip/hip_runtime.h>

// ---------- types ----------
typedef __attribute__((ext_vector_type(16))) __bf16 v16bf;
typedef __attribute__((ext_vector_type(8)))  float  v8f;
typedef __attribute__((ext_vector_type(8)))  unsigned short ushort8;

union Frag { ushort8 h[2]; v16bf v; };

__device__ __forceinline__ unsigned short f32_to_bf16(float x) {
    unsigned int u = __float_as_uint(x);
    return (unsigned short)((u + 0x7FFFu + ((u >> 16) & 1u)) >> 16);
}
__device__ __forceinline__ float silu(float x) { return x / (1.0f + __expf(-x)); }

// ---------- WMMA helper: one 16x16 C tile, accumulate over nkt K-tiles ----------
// arow: LDS pointer to this lane's A row (row m, K=0), 16B-aligned, K-contiguous bf16
// wfrag: swizzled weights, frag-major: [(ntile*nkt + kt)*32 + lane]*16 contiguous bf16
__device__ __forceinline__ v8f wmma_tiles(const unsigned short* arow,
                                          const unsigned short* wfrag,
                                          int nkt, int ntile, int khalf, int lane,
                                          v8f acc) {
    for (int kt = 0; kt < nkt; ++kt) {
        Frag A, B;
        const unsigned short* ka = arow + kt * 32 + khalf;
        A.h[0] = *(const ushort8*)(ka);
        A.h[1] = *(const ushort8*)(ka + 16);
        const ushort8* bp = (const ushort8*)(wfrag + (((size_t)ntile * nkt + kt) * 32 + lane) * 16);
        B.h[0] = bp[0];
        B.h[1] = bp[1];
        acc = __builtin_amdgcn_wmma_f32_16x16x32_bf16(false, A.v, false, B.v,
                                                      (short)0, acc, false, false);
    }
    return acc;
}

// ---------- weight prep: f32 (K,N) row-major -> bf16 swizzled fragments ----------
__global__ void prep_weights(const float* __restrict__ W, unsigned short* __restrict__ out,
                             int K, int Ncols) {
    int nkt = K / 32;
    int total = K * Ncols;
    int idx = blockIdx.x * blockDim.x + threadIdx.x;
    if (idx >= total) return;
    int e      = idx & 15;
    int lane   = (idx >> 4) & 31;
    int fragid = idx >> 9;
    int kt = fragid % nkt;
    int nt = fragid / nkt;
    int n = nt * 16 + (lane & 15);
    int khalf = ((lane >> 4) & 1) * 8;
    int k = kt * 32 + (e >> 3) * 16 + khalf + (e & 7);
    out[idx] = f32_to_bf16(W[(size_t)k * Ncols + n]);
}

__global__ void zero_f32(float* __restrict__ p, int n) {
    int i = blockIdx.x * blockDim.x + threadIdx.x;
    if (i < n) p[i] = 0.0f;
}

// ---------- edge phase: per-edge MLP (672->128->128->288) + scatter ----------
__global__ __launch_bounds__(128) void edge_kernel(
    const float* __restrict__ f, const float* __restrict__ s,
    const int* __restrict__ eidx, const float* __restrict__ ef, const float* __restrict__ es,
    const unsigned short* __restrict__ W1f, const float* __restrict__ b1,
    const unsigned short* __restrict__ W2f, const float* __restrict__ b2,
    const unsigned short* __restrict__ W3f, const float* __restrict__ b3,
    float* __restrict__ sum_fc, float* __restrict__ sum_sc, float* __restrict__ cnt,
    int Nn, int Ee) {
    __shared__ __align__(16) float          fbuf[4][16][60];   // _f: 16 edges x 3 x 20
    __shared__ __align__(16) unsigned short xbuf[4][16][672];  // MLP1 input, bf16
    __shared__ __align__(16) unsigned short h1b[4][16 * 128];
    __shared__ __align__(16) unsigned short h2b[4][16 * 128];
    __shared__ int sidx[4][16];
    __shared__ int didx[4][16];

    const int w = threadIdx.x >> 5, lane = threadIdx.x & 31;
    const int tile = blockIdx.x * 4 + w;
    const int e0 = tile * 16;
    if (e0 >= Ee) return;

    if (lane < 16) {
        int e = min(e0 + lane, Ee - 1);
        sidx[w][lane] = eidx[e];
        didx[w][lane] = eidx[Ee + e];
    }

    // build _f = concat(f[src], f[dst], edge_f): 16 x 3 x 20, float4 granularity
    // per edge: 6 chunks f[src] (3 rows x 2), 6 chunks f[dst], 3 chunks edge_f
    for (int idx = lane; idx < 16 * 15; idx += 32) {
        int e = idx / 15, r = idx % 15;
        float4 v;
        int off;  // element offset within fbuf row (multiple of 4)
        if (r < 6) {
            int i = r >> 1, hf = r & 1;
            v = ((const float4*)f)[(size_t)sidx[w][e] * 6 + i * 2 + hf];
            off = i * 20 + hf * 4;
        } else if (r < 12) {
            int rr = r - 6;
            int i = rr >> 1, hf = rr & 1;
            v = ((const float4*)f)[(size_t)didx[w][e] * 6 + i * 2 + hf];
            off = i * 20 + 8 + hf * 4;
        } else {
            int i = r - 12;
            v = ((const float4*)ef)[(size_t)min(e0 + e, Ee - 1) * 3 + i];
            off = i * 20 + 16;
        }
        *(float4*)&fbuf[w][e][off] = v;
    }

    // Gram f^T f (20x20) per edge, L2-normalized -> xbuf[:, 0:400] (two lanes per edge)
    {
        int e = lane & 15, half = lane >> 4;
        const float* fe = &fbuf[w][e][0];
        float ss = 0.0f;
        for (int idx = half * 200; idx < half * 200 + 200; ++idx) {
            int dc = idx / 20, kc = idx % 20;
            float g = fe[dc] * fe[kc] + fe[20 + dc] * fe[20 + kc] + fe[40 + dc] * fe[40 + kc];
            ss += g * g;
        }
        ss += __shfl_xor(ss, 16, 32);
        float inv = 1.0f / fmaxf(sqrtf(ss), 1e-12f);
        for (int idx = half * 200; idx < half * 200 + 200; ++idx) {
            int dc = idx / 20, kc = idx % 20;
            float g = fe[dc] * fe[kc] + fe[20 + dc] * fe[20 + kc] + fe[40 + dc] * fe[40 + kc];
            xbuf[w][e][idx] = f32_to_bf16(g * inv);
        }
    }

    // gather _s = concat(s[src], s[dst], edge_s) -> xbuf[:, 400:672], float4 granularity
    for (int idx = lane; idx < 16 * 68; idx += 32) {
        int e = idx / 68, c4 = idx % 68;
        float4 v;
        if (c4 < 32)      v = ((const float4*)s)[(size_t)sidx[w][e] * 32 + c4];
        else if (c4 < 64) v = ((const float4*)s)[(size_t)didx[w][e] * 32 + (c4 - 32)];
        else              v = ((const float4*)es)[(size_t)min(e0 + e, Ee - 1) * 4 + (c4 - 64)];
        unsigned short* xp = &xbuf[w][e][400 + c4 * 4];
        xp[0] = f32_to_bf16(v.x);
        xp[1] = f32_to_bf16(v.y);
        xp[2] = f32_to_bf16(v.z);
        xp[3] = f32_to_bf16(v.w);
    }

    const int m     = lane & 15;
    const int khalf = ((lane >> 4) & 1) * 8;
    const int mbase = ((lane >> 4) & 1) * 8;
    const int ncl   = lane & 15;

    // layer 1: 672 -> 128, silu
    for (int nt = 0; nt < 8; ++nt) {
        float bv = b1[nt * 16 + ncl];
        v8f acc = {bv, bv, bv, bv, bv, bv, bv, bv};
        acc = wmma_tiles(&xbuf[w][m][0], W1f, 21, nt, khalf, lane, acc);
#pragma unroll
        for (int v = 0; v < 8; ++v)
            h1b[w][(mbase + v) * 128 + nt * 16 + ncl] = f32_to_bf16(silu(acc[v]));
    }
    // layer 2: 128 -> 128, silu
    for (int nt = 0; nt < 8; ++nt) {
        float bv = b2[nt * 16 + ncl];
        v8f acc = {bv, bv, bv, bv, bv, bv, bv, bv};
        acc = wmma_tiles(&h1b[w][m * 128], W2f, 4, nt, khalf, lane, acc);
#pragma unroll
        for (int v = 0; v < 8; ++v)
            h2b[w][(mbase + v) * 128 + nt * 16 + ncl] = f32_to_bf16(silu(acc[v]));
    }
    // layer 3: 128 -> 288; cols [0,160) -> f_c weights in LDS, [160,288) -> s_c scatter
    float* fcw = (float*)&xbuf[w][0][0];  // reuse xbuf as 16 x 160 f32
    for (int nt = 0; nt < 18; ++nt) {
        float bv = b3[nt * 16 + ncl];
        v8f acc = {bv, bv, bv, bv, bv, bv, bv, bv};
        acc = wmma_tiles(&h2b[w][m * 128], W3f, 4, nt, khalf, lane, acc);
        int col = nt * 16 + ncl;
        if (col < 160) {
#pragma unroll
            for (int v = 0; v < 8; ++v) fcw[(mbase + v) * 160 + col] = acc[v];
        } else {
#pragma unroll
            for (int v = 0; v < 8; ++v) {
                int e = mbase + v;
                if (e0 + e < Ee)
                    atomicAdd(&sum_sc[(size_t)sidx[w][e] * 128 + (col - 160)], acc[v]);
            }
        }
    }

    // f_c = _f (3x20) @ fcw (20x8) per edge, scatter-add into sum_fc
    for (int idx = lane; idx < 384; idx += 32) {
        int e = idx / 24, r = idx % 24;
        int i = r / 8, k = r % 8;
        const float* fe = &fbuf[w][e][0];
        const float* cw = &fcw[e * 160];
        float acc = 0.0f;
        for (int d = 0; d < 20; ++d) acc += fe[i * 20 + d] * cw[d * 8 + k];
        if (e0 + e < Ee) atomicAdd(&sum_fc[(size_t)sidx[w][e] * 24 + i * 8 + k], acc);
    }
    if (lane < 16 && e0 + lane < Ee) atomicAdd(&cnt[sidx[w][lane]], 1.0f);
}

// ---------- node phase: per-node MLP (512->128->128->256) + outputs ----------
__global__ __launch_bounds__(128) void node_kernel(
    const float* __restrict__ f, const float* __restrict__ s,
    const unsigned short* __restrict__ V1f, const float* __restrict__ c1,
    const unsigned short* __restrict__ V2f, const float* __restrict__ c2,
    const unsigned short* __restrict__ V3f, const float* __restrict__ c3,
    const float* __restrict__ sum_fc, const float* __restrict__ sum_sc,
    const float* __restrict__ cnt, float* __restrict__ out, int Nn) {
    __shared__ __align__(16) float          tf[4][16][48];     // temp_f: 16 nodes x 3 x 16
    __shared__ __align__(16) unsigned short xbuf[4][16][512];
    __shared__ __align__(16) unsigned short h1b[4][16 * 128];
    __shared__ __align__(16) unsigned short h2b[4][16 * 128];

    const int w = threadIdx.x >> 5, lane = threadIdx.x & 31;
    const int tile = blockIdx.x * 4 + w;
    const int n0 = tile * 16;
    if (n0 >= Nn) return;

    // temp_f = concat(f, f_c_mean): 16 x 3 x 16, float4 granularity
    // per node: 6 chunks of f (3 rows x 2), 6 chunks of f_c_mean
    for (int idx = lane; idx < 16 * 12; idx += 32) {
        int e = idx / 12, r = idx % 12;
        int n = min(n0 + e, Nn - 1);
        float4 v;
        int off;
        if (r < 6) {
            int i = r >> 1, hf = r & 1;
            v = ((const float4*)f)[(size_t)n * 6 + i * 2 + hf];
            off = i * 16 + hf * 4;
        } else {
            int rr = r - 6;
            int i = rr >> 1, hf = rr & 1;
            v = ((const float4*)sum_fc)[(size_t)n * 6 + i * 2 + hf];
            float ic = 1.0f / fmaxf(cnt[n], 1.0f);
            v.x *= ic; v.y *= ic; v.z *= ic; v.w *= ic;
            off = i * 16 + 8 + hf * 4;
        }
        *(float4*)&tf[w][e][off] = v;
    }

    // Gram (16x16) per node, normalized -> xbuf[:, 0:256]
    {
        int e = lane & 15, half = lane >> 4;
        const float* fe = &tf[w][e][0];
        float ss = 0.0f;
        for (int idx = half * 128; idx < half * 128 + 128; ++idx) {
            int dc = idx >> 4, kc = idx & 15;
            float g = fe[dc] * fe[kc] + fe[16 + dc] * fe[16 + kc] + fe[32 + dc] * fe[32 + kc];
            ss += g * g;
        }
        ss += __shfl_xor(ss, 16, 32);
        float inv = 1.0f / fmaxf(sqrtf(ss), 1e-12f);
        for (int idx = half * 128; idx < half * 128 + 128; ++idx) {
            int dc = idx >> 4, kc = idx & 15;
            float g = fe[dc] * fe[kc] + fe[16 + dc] * fe[16 + kc] + fe[32 + dc] * fe[32 + kc];
            xbuf[w][e][idx] = f32_to_bf16(g * inv);
        }
    }

    // concat(s, s_c_mean) -> xbuf[:, 256:512], float4 granularity
    for (int idx = lane; idx < 16 * 64; idx += 32) {
        int e = idx / 64, c4 = idx % 64;
        int n = min(n0 + e, Nn - 1);
        float4 v;
        if (c4 < 32) {
            v = ((const float4*)s)[(size_t)n * 32 + c4];
        } else {
            v = ((const float4*)sum_sc)[(size_t)n * 32 + (c4 - 32)];
            float ic = 1.0f / fmaxf(cnt[n], 1.0f);
            v.x *= ic; v.y *= ic; v.z *= ic; v.w *= ic;
        }
        unsigned short* xp = &xbuf[w][e][256 + c4 * 4];
        xp[0] = f32_to_bf16(v.x);
        xp[1] = f32_to_bf16(v.y);
        xp[2] = f32_to_bf16(v.z);
        xp[3] = f32_to_bf16(v.w);
    }

    const int m     = lane & 15;
    const int khalf = ((lane >> 4) & 1) * 8;
    const int mbase = ((lane >> 4) & 1) * 8;
    const int ncl   = lane & 15;

    for (int nt = 0; nt < 8; ++nt) {
        float bv = c1[nt * 16 + ncl];
        v8f acc = {bv, bv, bv, bv, bv, bv, bv, bv};
        acc = wmma_tiles(&xbuf[w][m][0], V1f, 16, nt, khalf, lane, acc);
#pragma unroll
        for (int v = 0; v < 8; ++v)
            h1b[w][(mbase + v) * 128 + nt * 16 + ncl] = f32_to_bf16(silu(acc[v]));
    }
    for (int nt = 0; nt < 8; ++nt) {
        float bv = c2[nt * 16 + ncl];
        v8f acc = {bv, bv, bv, bv, bv, bv, bv, bv};
        acc = wmma_tiles(&h1b[w][m * 128], V2f, 4, nt, khalf, lane, acc);
#pragma unroll
        for (int v = 0; v < 8; ++v)
            h2b[w][(mbase + v) * 128 + nt * 16 + ncl] = f32_to_bf16(silu(acc[v]));
    }
    // layer 3: 128 -> 256; cols [0,128) -> temp_f_c in LDS, [128,256) -> temp_s_c out
    float* tfc = (float*)&xbuf[w][0][0];  // reuse xbuf as 16 x 128 f32
    for (int nt = 0; nt < 16; ++nt) {
        float bv = c3[nt * 16 + ncl];
        v8f acc = {bv, bv, bv, bv, bv, bv, bv, bv};
        acc = wmma_tiles(&h2b[w][m * 128], V3f, 4, nt, khalf, lane, acc);
        int col = nt * 16 + ncl;
        if (col < 128) {
#pragma unroll
            for (int v = 0; v < 8; ++v) tfc[(mbase + v) * 128 + col] = acc[v];
        } else {
#pragma unroll
            for (int v = 0; v < 8; ++v) {
                int n = n0 + mbase + v;
                if (n < Nn)
                    out[(size_t)Nn * 24 + (size_t)n * 128 + (col - 128)] = acc[v];
            }
        }
    }

    // f_out = temp_f (3x16) @ temp_f_c (16x8)
    for (int idx = lane; idx < 384; idx += 32) {
        int e = idx / 24, r = idx % 24;
        int i = r / 8, k = r % 8;
        const float* fe = &tf[w][e][0];
        const float* cw = &tfc[e * 128];
        float acc = 0.0f;
        for (int d = 0; d < 16; ++d) acc += fe[i * 16 + d] * cw[d * 8 + k];
        int n = n0 + e;
        if (n < Nn) out[(size_t)n * 24 + i * 8 + k] = acc;
    }
}

// ---------- host launcher ----------
extern "C" void kernel_launch(void* const* d_in, const int* in_sizes, int n_in,
                              void* d_out, int out_size, void* d_ws, size_t ws_size,
                              hipStream_t stream) {
    const float* f  = (const float*)d_in[0];
    const float* s  = (const float*)d_in[1];
    const int* eidx = (const int*)d_in[2];
    const float* ef = (const float*)d_in[3];
    const float* es = (const float*)d_in[4];
    const float* W1 = (const float*)d_in[5];  const float* b1 = (const float*)d_in[6];
    const float* W2 = (const float*)d_in[7];  const float* b2 = (const float*)d_in[8];
    const float* W3 = (const float*)d_in[9];  const float* b3 = (const float*)d_in[10];
    const float* V1 = (const float*)d_in[11]; const float* c1 = (const float*)d_in[12];
    const float* V2 = (const float*)d_in[13]; const float* c2 = (const float*)d_in[14];
    const float* V3 = (const float*)d_in[15]; const float* c3 = (const float*)d_in[16];

    const int Nn = in_sizes[0] / 24;   // f: (N,3,8)
    const int Ee = in_sizes[3] / 12;   // edge_f: (E,3,4)

    // workspace layout
    char* p = (char*)d_ws;
    unsigned short* W1f = (unsigned short*)p; p += (size_t)672 * 128 * 2;
    unsigned short* W2f = (unsigned short*)p; p += (size_t)128 * 128 * 2;
    unsigned short* W3f = (unsigned short*)p; p += (size_t)128 * 288 * 2;
    unsigned short* V1f = (unsigned short*)p; p += (size_t)512 * 128 * 2;
    unsigned short* V2f = (unsigned short*)p; p += (size_t)128 * 128 * 2;
    unsigned short* V3f = (unsigned short*)p; p += (size_t)128 * 256 * 2;
    float* sum_fc = (float*)p; p += (size_t)Nn * 24 * 4;
    float* sum_sc = (float*)p; p += (size_t)Nn * 128 * 4;
    float* cnt    = (float*)p; p += (size_t)Nn * 4;

    prep_weights<<<(672 * 128 + 255) / 256, 256, 0, stream>>>(W1, W1f, 672, 128);
    prep_weights<<<(128 * 128 + 255) / 256, 256, 0, stream>>>(W2, W2f, 128, 128);
    prep_weights<<<(128 * 288 + 255) / 256, 256, 0, stream>>>(W3, W3f, 128, 288);
    prep_weights<<<(512 * 128 + 255) / 256, 256, 0, stream>>>(V1, V1f, 512, 128);
    prep_weights<<<(128 * 128 + 255) / 256, 256, 0, stream>>>(V2, V2f, 128, 128);
    prep_weights<<<(128 * 256 + 255) / 256, 256, 0, stream>>>(V3, V3f, 128, 256);

    int ztot = Nn * (24 + 128 + 1);  // sum_fc, sum_sc, cnt are contiguous
    zero_f32<<<(ztot + 255) / 256, 256, 0, stream>>>(sum_fc, ztot);

    int etiles = (Ee + 15) / 16;
    edge_kernel<<<(etiles + 3) / 4, 128, 0, stream>>>(
        f, s, eidx, ef, es, W1f, b1, W2f, b2, W3f, b3, sum_fc, sum_sc, cnt, Nn, Ee);

    int ntiles = (Nn + 15) / 16;
    node_kernel<<<(ntiles + 3) / 4, 128, 0, stream>>>(
        f, s, V1f, c1, V2f, c2, V3f, c3, sum_fc, sum_sc, cnt, (float*)d_out, Nn);
}
